// GATv2Model_36421322670669
// MI455X (gfx1250) — matrix operand order repaired and checked by reading, not compiled, not run on hardware
//
#include <hip/hip_runtime.h>

#define NNODES 30000
#define NEDGES 300000
#define ETOT   (NEDGES + NNODES)   // 330000, self loops appended
#define NGRAPH 128
#define INDIM  128
#define HIDC   64
#define NHEADS 4
#define CFULL  256                 // HIDC * NHEADS
#define NCLS   10
#define BN_EPS 1e-5f
#define SLOPE  0.2f

typedef __attribute__((ext_vector_type(16))) __bf16 v16bf;
typedef __attribute__((ext_vector_type(8)))  float  v8f;

// ---------- helpers ----------

__device__ __forceinline__ unsigned short f2bf(float f) {
    unsigned u = __float_as_uint(f);
    u += 0x7FFFu + ((u >> 16) & 1u);      // round-to-nearest-even
    return (unsigned short)(u >> 16);
}

// monotone float -> uint mapping so atomicMax(uint) == float max
__device__ __forceinline__ unsigned fenc(float f) {
    unsigned u = __float_as_uint(f);
    return (u & 0x80000000u) ? ~u : (u | 0x80000000u);
}
__device__ __forceinline__ float fdec(unsigned k) {
    unsigned u = (k & 0x80000000u) ? (k ^ 0x80000000u) : ~k;
    return __uint_as_float(u);
}

__device__ __forceinline__ void edge_sd(const int* __restrict__ ei, int eid,
                                        int& s, int& d) {
    if (eid < NEDGES) { s = ei[eid]; d = ei[NEDGES + eid]; }
    else              { s = d = eid - NEDGES; }   // self loop
}

// ---------- conversions ----------

__global__ void f32_to_bf16_k(const float* __restrict__ src,
                              unsigned short* __restrict__ dst, int n) {
    int t = blockIdx.x * blockDim.x + threadIdx.x;
    if (t < n) dst[t] = f2bf(src[t]);
}

// W[K][256] fp32 -> Wt[256][K] bf16 (so B fragments load contiguously in K)
__global__ void wT_bf16_k(const float* __restrict__ W,
                          unsigned short* __restrict__ Wt, int K) {
    int t = blockIdx.x * blockDim.x + threadIdx.x;
    if (t >= K * CFULL) return;
    int k = t / CFULL, n = t % CFULL;
    Wt[(size_t)n * K + k] = f2bf(W[t]);
}

// ---------- WMMA GEMM: C[M x 256] = A[M x K](bf16) * Wt^T, K % 32 == 0 ----------
// grid = (M/16), block = 128 (4 waves). Each wave owns a 16x64 output strip:
// one A fragment per k-step is reused across 4 WMMAs (4x less A traffic).

union Frag { v16bf v; uint4 q[2]; };

__global__ __launch_bounds__(128)
void gemm_bf16_wmma(const unsigned short* __restrict__ A,
                    const unsigned short* __restrict__ Bt,
                    float* __restrict__ C, int K) {
    const int lane    = threadIdx.x & 31;
    const int wave    = threadIdx.x >> 5;
    const int hi      = lane >> 4;     // half-wave select
    const int mr      = lane & 15;
    const int row0    = blockIdx.x * 16;
    const int colBase = wave * 64;

    // A fragment: lane holds row (row0+mr); elements 0..7 -> k = kk+hi*8+e,
    // elements 8..15 -> k = kk+16+hi*8+(e-8)   (ISA 16-bit A 16x32 layout)
    const unsigned short* pa = A  + (size_t)(row0 + mr) * K + hi * 8;
    // B fragment (from transposed weights): lane holds col; tile j adds 16*K.
    // elements 0..7 -> k = kk + hi*8 + e, 8..15 -> k = kk + 16 + hi*8 + (e-8)
    const unsigned short* pb = Bt + (size_t)(colBase + mr) * K + hi * 8;

    v8f acc[4] = {{}, {}, {}, {}};
    for (int kk = 0; kk < K; kk += 32) {
        Frag a;
        a.q[0] = *reinterpret_cast<const uint4*>(pa + kk);
        a.q[1] = *reinterpret_cast<const uint4*>(pa + kk + 16);
        #pragma unroll
        for (int j = 0; j < 4; ++j) {
            Frag b;
            const unsigned short* pbj = pb + (size_t)j * 16 * K + kk;
            b.q[0] = *reinterpret_cast<const uint4*>(pbj);
            b.q[1] = *reinterpret_cast<const uint4*>(pbj + 16);
            acc[j] = __builtin_amdgcn_wmma_f32_16x16x32_bf16(
                         false, a.v, false, b.v, (short)0, acc[j], false, false);
        }
    }
    // D layout: VGPR r -> M = r (lanes 0-15) / r+8 (lanes 16-31), N = mr
    float* pc = C + (size_t)(row0 + hi * 8) * CFULL + colBase + mr;
    #pragma unroll
    for (int j = 0; j < 4; ++j) {
        #pragma unroll
        for (int r = 0; r < 8; ++r) pc[(size_t)r * CFULL + j * 16] = acc[j][r];
    }
}

// ---------- edge kernels (GATv2 attention) ----------

__global__ void edge_score_k(const float* __restrict__ xl,
                             const float* __restrict__ xr,
                             const float* __restrict__ att,
                             const int* __restrict__ ei,
                             float* __restrict__ eb,
                             unsigned* __restrict__ mx) {
    int t = blockIdx.x * blockDim.x + threadIdx.x;
    if (t >= ETOT * NHEADS) return;
    int eid = t >> 2, hh = t & 3;
    int s, d; edge_sd(ei, eid, s, d);
    const float4* pl = (const float4*)(xl + (size_t)s * CFULL + hh * HIDC);
    const float4* pr = (const float4*)(xr + (size_t)d * CFULL + hh * HIDC);
    const float4* pw = (const float4*)(att + hh * HIDC);
    float acc = 0.f;
    #pragma unroll 4
    for (int c = 0; c < HIDC / 4; ++c) {
        float4 a = pl[c], b = pr[c], w = pw[c];
        float z;
        z = a.x + b.x; z = z > 0.f ? z : SLOPE * z; acc += w.x * z;
        z = a.y + b.y; z = z > 0.f ? z : SLOPE * z; acc += w.y * z;
        z = a.z + b.z; z = z > 0.f ? z : SLOPE * z; acc += w.z * z;
        z = a.w + b.w; z = z > 0.f ? z : SLOPE * z; acc += w.w * z;
    }
    eb[t] = acc;
    atomicMax(&mx[d * NHEADS + hh], fenc(acc));
}

__global__ void edge_exp_k(const int* __restrict__ ei,
                           float* __restrict__ eb,
                           const unsigned* __restrict__ mx,
                           float* __restrict__ sm) {
    int t = blockIdx.x * blockDim.x + threadIdx.x;
    if (t >= ETOT * NHEADS) return;
    int eid = t >> 2, hh = t & 3;
    int s, d; edge_sd(ei, eid, s, d);
    float p = expf(eb[t] - fdec(mx[d * NHEADS + hh]));
    eb[t] = p;
    atomicAdd(&sm[d * NHEADS + hh], p);
}

__global__ void edge_aggr_k(const int* __restrict__ ei,
                            const float* __restrict__ eb,
                            const float* __restrict__ sm,
                            const float* __restrict__ xl,
                            float* __restrict__ out, int outC, int concat) {
    int t = blockIdx.x * blockDim.x + threadIdx.x;
    if (t >= ETOT * NHEADS) return;
    int eid = t >> 2, hh = t & 3;
    int s, d; edge_sd(ei, eid, s, d);
    float alpha = eb[t] / sm[d * NHEADS + hh];
    if (!concat) alpha *= 0.25f;                 // mean over heads
    const float* pl = xl + (size_t)s * CFULL + hh * HIDC;
    float* po = out + (size_t)d * outC + (concat ? hh * HIDC : 0);
    #pragma unroll 4
    for (int c = 0; c < HIDC; ++c) atomicAdd(po + c, alpha * pl[c]);
}

// ---------- batchnorm (train stats, biased var) + ELU ----------

__global__ void bn_stats_k(const float* __restrict__ x, int C,
                           float* __restrict__ mu, float* __restrict__ rv) {
    int c = blockIdx.x;
    float s = 0.f, s2 = 0.f;
    for (int i = threadIdx.x; i < NNODES; i += blockDim.x) {
        float v = x[(size_t)i * C + c];
        s += v; s2 += v * v;
    }
    __shared__ float sh[256], sh2[256];
    sh[threadIdx.x] = s; sh2[threadIdx.x] = s2;
    __syncthreads();
    for (int off = 128; off; off >>= 1) {
        if ((int)threadIdx.x < off) {
            sh[threadIdx.x]  += sh[threadIdx.x + off];
            sh2[threadIdx.x] += sh2[threadIdx.x + off];
        }
        __syncthreads();
    }
    if (threadIdx.x == 0) {
        float m   = sh[0] / (float)NNODES;
        float var = sh2[0] / (float)NNODES - m * m;
        mu[c] = m;
        rv[c] = rsqrtf(var + BN_EPS);
    }
}

__global__ void bn_elu_k(float* __restrict__ h, const float* __restrict__ mu,
                         const float* __restrict__ rv, const float* __restrict__ g,
                         const float* __restrict__ bb, int cmask, int n) {
    int t = blockIdx.x * blockDim.x + threadIdx.x;
    if (t >= n) return;
    int c = t & cmask;                            // C is a power of two
    float v = (h[t] - mu[c]) * rv[c] * g[c] + bb[c];
    h[t] = v > 0.f ? v : expm1f(v);               // ELU(alpha=1)
}

// ---------- global mean pool + MLP head ----------

__global__ void pool_k(const float* __restrict__ h, const int* __restrict__ batch,
                       float* __restrict__ pooled, float* __restrict__ cnt) {
    int t = blockIdx.x * blockDim.x + threadIdx.x;
    if (t >= NNODES * HIDC) return;
    int i = t >> 6, c = t & 63;
    int b = batch[i];
    atomicAdd(&pooled[b * HIDC + c], h[t]);
    if (c == 0) atomicAdd(&cnt[b], 1.f);
}

__global__ void head_k(const float* __restrict__ pooled, const float* __restrict__ cnt,
                       const float* __restrict__ w1, const float* __restrict__ b1,
                       const float* __restrict__ w2, const float* __restrict__ b2,
                       float* __restrict__ out) {
    int g = threadIdx.x;
    if (g >= NGRAPH) return;
    float inv = 1.f / fmaxf(cnt[g], 1.f);
    float p[HIDC];
    #pragma unroll
    for (int k = 0; k < HIDC; ++k) p[k] = pooled[g * HIDC + k] * inv;
    float acc[NCLS];
    #pragma unroll
    for (int n = 0; n < NCLS; ++n) acc[n] = b2[n];
    for (int j = 0; j < HIDC; ++j) {
        float hv = b1[j];
        #pragma unroll 8
        for (int k = 0; k < HIDC; ++k) hv += p[k] * w1[k * HIDC + j];
        hv = fmaxf(hv, 0.f);
        #pragma unroll
        for (int n = 0; n < NCLS; ++n) acc[n] += hv * w2[j * NCLS + n];
    }
    #pragma unroll
    for (int n = 0; n < NCLS; ++n) out[g * NCLS + n] = acc[n];
}

// ---------- host ----------

extern "C" void kernel_launch(void* const* d_in, const int* in_sizes, int n_in,
                              void* d_out, int out_size, void* d_ws, size_t ws_size,
                              hipStream_t stream) {
    (void)in_sizes; (void)n_in; (void)out_size; (void)ws_size;

    const float* x     = (const float*)d_in[0];
    const int*   ei    = (const int*)d_in[1];
    const int*   batch = (const int*)d_in[2];
    const float* fc1w  = (const float*)d_in[27];
    const float* fc1b  = (const float*)d_in[28];
    const float* fc2w  = (const float*)d_in[29];
    const float* fc2b  = (const float*)d_in[30];

    char* ws = (char*)d_ws;
    size_t off = 0;
    auto take = [&](size_t bytes) -> char* {
        char* p = ws + off;
        off += (bytes + 255) & ~(size_t)255;
        return p;
    };
    float*          h      = (float*)take((size_t)NNODES * CFULL * 4);
    unsigned short* hb     = (unsigned short*)take((size_t)NNODES * CFULL * 2);
    unsigned short* wtl    = (unsigned short*)take((size_t)CFULL * CFULL * 2);
    unsigned short* wtr    = (unsigned short*)take((size_t)CFULL * CFULL * 2);
    float*          xl     = (float*)take((size_t)NNODES * CFULL * 4);
    float*          xr     = (float*)take((size_t)NNODES * CFULL * 4);
    float*          eb     = (float*)take((size_t)ETOT * NHEADS * 4);
    unsigned*       mx     = (unsigned*)take((size_t)NNODES * NHEADS * 4);
    float*          sm     = (float*)take((size_t)NNODES * NHEADS * 4);
    float*          mu     = (float*)take(CFULL * 4);
    float*          rv     = (float*)take(CFULL * 4);
    float*          pooled = (float*)take((size_t)NGRAPH * HIDC * 4);
    float*          cnt    = (float*)take((size_t)NGRAPH * 4);

    const int ETH     = ETOT * NHEADS;
    const int eblocks = (ETH + 255) / 256;
    const int inC[4]  = {INDIM, CFULL, CFULL, CFULL};

    for (int L = 0; L < 4; ++L) {
        const float* Wl  = (const float*)d_in[3 + 6 * L];
        const float* Wr  = (const float*)d_in[4 + 6 * L];
        const float* att = (const float*)d_in[5 + 6 * L];
        // bias b{L} (d_in[6+6L]) is algebraically cancelled by BatchNorm
        const float* gn  = (const float*)d_in[7 + 6 * L];
        const float* bb  = (const float*)d_in[8 + 6 * L];
        const int K      = inC[L];
        const int outC   = (L < 3) ? CFULL : HIDC;
        const int concat = (L < 3) ? 1 : 0;
        const float* hin = (L == 0) ? x : h;

        int nh = NNODES * K;
        f32_to_bf16_k<<<(nh + 255) / 256, 256, 0, stream>>>(hin, hb, nh);
        wT_bf16_k<<<(K * CFULL + 255) / 256, 256, 0, stream>>>(Wl, wtl, K);
        wT_bf16_k<<<(K * CFULL + 255) / 256, 256, 0, stream>>>(Wr, wtr, K);

        dim3 gg(NNODES / 16);                 // 30000 = 1875*16
        gemm_bf16_wmma<<<gg, 128, 0, stream>>>(hb, wtl, xl, K);
        gemm_bf16_wmma<<<gg, 128, 0, stream>>>(hb, wtr, xr, K);

        hipMemsetAsync(h,  0, (size_t)NNODES * outC * 4, stream);
        hipMemsetAsync(mx, 0, (size_t)NNODES * NHEADS * 4, stream);
        hipMemsetAsync(sm, 0, (size_t)NNODES * NHEADS * 4, stream);

        edge_score_k<<<eblocks, 256, 0, stream>>>(xl, xr, att, ei, eb, mx);
        edge_exp_k  <<<eblocks, 256, 0, stream>>>(ei, eb, mx, sm);
        edge_aggr_k <<<eblocks, 256, 0, stream>>>(ei, eb, sm, xl, h, outC, concat);

        bn_stats_k<<<outC, 256, 0, stream>>>(h, outC, mu, rv);
        int ne = NNODES * outC;
        bn_elu_k<<<(ne + 255) / 256, 256, 0, stream>>>(h, mu, rv, gn, bb, outC - 1, ne);
    }

    hipMemsetAsync(pooled, 0, (size_t)NGRAPH * HIDC * 4, stream);
    hipMemsetAsync(cnt,    0, (size_t)NGRAPH * 4, stream);
    pool_k<<<(NNODES * HIDC + 255) / 256, 256, 0, stream>>>(h, batch, pooled, cnt);
    head_k<<<1, 128, 0, stream>>>(pooled, cnt, fc1w, fc1b, fc2w, fc2b, (float*)d_out);
}
